// INLCachedAttention_3599182594269
// MI455X (gfx1250) — compile-verified, matrix operand order checked by reading, not run on hardware
//
#include <hip/hip_runtime.h>
#include <cstdint>

// ---------------------------------------------------------------------------
// MI455X (gfx1250) causal MHA: bf16 WMMA everywhere, f32 accumulate.
// B=2, S=2048, E=2048, H=16, D=128
// GEMMs use a 64x64-per-wave register tile: 16 v_wmma per K=32 step against
// 16 global_load_b128 (32 FLOP/byte from L2; whole working set is L2-resident).
// ---------------------------------------------------------------------------

#define B_ 2
#define S_ 2048
#define E_ 2048
#define H_ 16
#define D_ 128
#define M_ (B_ * S_)      // 4096 rows of x / attn_out
#define N3_ (3 * E_)      // 6144

typedef __bf16 bf16x16 __attribute__((ext_vector_type(16)));
typedef float  f32x8   __attribute__((ext_vector_type(8)));

union Frag {            // one 16x32 bf16 WMMA operand fragment (per lane: 32B)
  bf16x16  v;
  uint32_t u[8];
  uint4    q[2];
};
union FAcc {            // one 16x16 f32 C/D fragment (per lane: 8 floats)
  f32x8 v;
  float f[8];
};

__device__ __forceinline__ uint16_t f2bf(float x) {
  uint32_t u = __float_as_uint(x);
  uint32_t r = (u + 0x7FFFu + ((u >> 16) & 1u)) >> 16;   // round-to-nearest-even
  return (uint16_t)r;
}
__device__ __forceinline__ uint32_t pack_bf2(float lo, float hi) {
  return (uint32_t)f2bf(lo) | ((uint32_t)f2bf(hi) << 16);
}
__device__ __forceinline__ void load_frag(Frag& f, const uint16_t* p) {
  f.q[0] = *(const uint4*)(p);        // K-offsets {ko..ko+7}
  f.q[1] = *(const uint4*)(p + 16);   // K-offsets {ko+16..ko+23}
}

#define WMMA_BF16(A, Bm, C) \
  __builtin_amdgcn_wmma_f32_16x16x32_bf16(false, (A), false, (Bm), (short)0, (C), false, false)

// ---------------------------------------------------------------------------
// Kernel 0: f32 -> bf16 conversion (memory bound, negligible cost)
// ---------------------------------------------------------------------------
__global__ void cvt_f32_bf16(const float* __restrict__ src,
                             uint16_t* __restrict__ dst, int n) {
  int i = (blockIdx.x * blockDim.x + threadIdx.x) * 4;
  if (i + 3 < n) {
    float4 f = *(const float4*)(src + i);
    uint2 o;
    o.x = pack_bf2(f.x, f.y);
    o.y = pack_bf2(f.z, f.w);
    *(uint2*)(dst + i) = o;
  }
}

// ---------------------------------------------------------------------------
// Shared 64x64-per-wave bf16 GEMM main loop: C = A[M,K] * W^T, W is [N,K]
// row-major, so both operands use the identical row-pattern fragment load.
// 16 accumulators (128 VGPRs), 4 A-frags + 4 B-frags per K=32 step.
// ---------------------------------------------------------------------------
__device__ __forceinline__ void gemm64x64(const uint16_t* __restrict__ A, int lda,
                                          const uint16_t* __restrict__ W, int ldb,
                                          int K, int mbase, int nbase, int lane,
                                          f32x8 acc[16]) {
  const int r15 = lane & 15;
  const int ko  = (lane < 16) ? 0 : 8;
  const uint16_t* pa[4];
  const uint16_t* pb[4];
#pragma unroll
  for (int t = 0; t < 4; ++t) {
    pa[t] = A + (size_t)(mbase + 16 * t + r15) * lda + ko;
    pb[t] = W + (size_t)(nbase + 16 * t + r15) * ldb + ko;
  }

  const f32x8 z = {0.f, 0.f, 0.f, 0.f, 0.f, 0.f, 0.f, 0.f};
#pragma unroll
  for (int t = 0; t < 16; ++t) acc[t] = z;

  for (int k0 = 0; k0 < K; k0 += 32) {
    Frag a[4], b[4];
#pragma unroll
    for (int t = 0; t < 4; ++t) load_frag(a[t], pa[t] + k0);
#pragma unroll
    for (int t = 0; t < 4; ++t) load_frag(b[t], pb[t] + k0);
    __builtin_prefetch(pa[0] + k0 + 256, 0, 1);   // global_prefetch_b8
    __builtin_prefetch(pb[0] + k0 + 256, 0, 1);
#pragma unroll
    for (int i = 0; i < 4; ++i)
#pragma unroll
      for (int j = 0; j < 4; ++j)
        acc[i * 4 + j] = WMMA_BF16(a[i].v, b[j].v, acc[i * 4 + j]);
  }
}

// ---------------------------------------------------------------------------
// Kernel 1: QKV GEMM + bias; scatter Q,K -> [B,H,S,D] bf16, V -> [B,H,D,S] bf16
// ---------------------------------------------------------------------------
__global__ __launch_bounds__(128)
void qkv_gemm_kernel(const uint16_t* __restrict__ X,   // [M_, E_] bf16
                     const uint16_t* __restrict__ W,   // [N3_, E_] bf16
                     const float* __restrict__ bias,   // [N3_]
                     uint16_t* __restrict__ Qo,        // [B,H,S,D]
                     uint16_t* __restrict__ Ko,        // [B,H,S,D]
                     uint16_t* __restrict__ Vt) {      // [B,H,D,S]
  const int lane = threadIdx.x & 31;
  const int wave = threadIdx.x >> 5;
  const int bm = blockIdx.x & 31;          // M_/128 = 32
  const int bn = blockIdx.x >> 5;          // N3_/128 = 48
  const int mbase = bm * 128 + (wave & 1) * 64;
  const int nbase = bn * 128 + (wave >> 1) * 64;

  f32x8 acc[16];
  gemm64x64(X, E_, W, E_, E_, mbase, nbase, lane, acc);

  const int r15 = lane & 15;
  const int mo  = (lane < 16) ? 0 : 8;     // C layout: vgpr r -> M = r (+8 hi half)
#pragma unroll
  for (int j = 0; j < 4; ++j) {
    const int ncol  = nbase + 16 * j + r15;
    const float bj  = bias[ncol];
    const int which = ncol >> 11;          // 0=Q 1=K 2=V
    const int rem   = ncol & (E_ - 1);
    const int hh    = rem >> 7;            // head
    const int dd    = rem & (D_ - 1);      // dim
#pragma unroll
    for (int i = 0; i < 4; ++i) {
      FAcc fa; fa.v = acc[i * 4 + j];
#pragma unroll
      for (int r = 0; r < 8; ++r) {
        const int mm  = mbase + 16 * i + r + mo;
        const int bb  = mm >> 11;
        const int ss  = mm & (S_ - 1);
        const int bhh = (bb << 4) + hh;
        const uint16_t hv = f2bf(fa.f[r] + bj);
        if (which == 0)      Qo[((size_t)bhh * S_ + ss) * D_ + dd] = hv;
        else if (which == 1) Ko[((size_t)bhh * S_ + ss) * D_ + dd] = hv;
        else                 Vt[((size_t)bhh * D_ + dd) * S_ + ss] = hv;
      }
    }
  }
}

// ---------------------------------------------------------------------------
// Kernel 2: flash-style causal attention. One 16-query tile per wave.
// Trick: compute S^T = K * Q^T so (a) the softmax axis is per-lane, and
// (b) the two 16-key score C-fragments pack *directly* into the P A-fragment
// layout for P*V (lane<16 holds K={0..7,16..23}, lane>=16 K={8..15,24..31}).
// V is pre-transposed, so the P*V B-fragment is a plain row-pattern load.
// ---------------------------------------------------------------------------
__global__ __launch_bounds__(256)
void attn_kernel(const uint16_t* __restrict__ Qb,
                 const uint16_t* __restrict__ Kb,
                 const uint16_t* __restrict__ Vtb,
                 uint16_t* __restrict__ Ob) {          // [M_, E_] bf16
  const int lane = threadIdx.x & 31;
  const int wave = threadIdx.x >> 5;
  const int qchunks = S_ / 128;                        // 16
  const int bh = blockIdx.x / qchunks;
  const int qc = blockIdx.x % qchunks;
  const int q0 = qc * 128 + wave * 16;
  const int b  = bh >> 4, h = bh & (H_ - 1);

  const uint16_t* qp = Qb  + (size_t)bh * S_ * D_;
  const uint16_t* kp = Kb  + (size_t)bh * S_ * D_;
  const uint16_t* vp = Vtb + (size_t)bh * D_ * S_;

  const int r15 = lane & 15;
  const int ko  = (lane < 16) ? 0 : 8;
  const int qv  = q0 + r15;                 // query this lane owns in softmax state
  const float scale = 0.08838834764831843f; // 1/sqrt(128)

  Frag Qf[4];                               // Q tile as B-operand, 4 x (16x32 over D)
#pragma unroll
  for (int kd = 0; kd < 4; ++kd)
    load_frag(Qf[kd], qp + (size_t)(q0 + r15) * D_ + kd * 32 + ko);

  FAcc O[8];                                // 16 queries x 128 d, f32
#pragma unroll
  for (int n = 0; n < 8; ++n)
#pragma unroll
    for (int r = 0; r < 8; ++r) O[n].f[r] = 0.f;

  float mrun = -1e30f, lrun = 0.f;
  const int qsel = (lane < 16) ? 0 : 8;     // O vgpr r -> query r (+8 hi half)

  for (int kb = 0; kb <= q0 + 15; kb += 32) {
    // --- scores S^T for two 16-key tiles -------------------------------
    FAcc s0, s1;
    {
      f32x8 a = {0.f,0.f,0.f,0.f,0.f,0.f,0.f,0.f};
#pragma unroll
      for (int kd = 0; kd < 4; ++kd) {
        Frag kf; load_frag(kf, kp + (size_t)(kb + r15) * D_ + kd * 32 + ko);
        a = WMMA_BF16(kf.v, Qf[kd].v, a);
      }
      s0.v = a;
    }
    {
      f32x8 a = {0.f,0.f,0.f,0.f,0.f,0.f,0.f,0.f};
#pragma unroll
      for (int kd = 0; kd < 4; ++kd) {
        Frag kf; load_frag(kf, kp + (size_t)(kb + 16 + r15) * D_ + kd * 32 + ko);
        a = WMMA_BF16(kf.v, Qf[kd].v, a);
      }
      s1.v = a;
    }
    // --- scale + causal mask + online softmax --------------------------
    float cm = -1e30f;
#pragma unroll
    for (int r = 0; r < 8; ++r) {
      const int kv0 = kb + r + ko;          // key of s0 vgpr r on this lane
      const int kv1 = kv0 + 16;
      float v0 = s0.f[r] * scale; if (kv0 > qv) v0 = -1e30f;
      float v1 = s1.f[r] * scale; if (kv1 > qv) v1 = -1e30f;
      s0.f[r] = v0; s1.f[r] = v1;
      cm = fmaxf(cm, fmaxf(v0, v1));
    }
    cm = fmaxf(cm, __shfl_xor(cm, 16, 32)); // lane pair holds same query
    const float mnew  = fmaxf(mrun, cm);
    const float alpha = __expf(mrun - mnew);
    float p0[8], p1[8], cs = 0.f;
#pragma unroll
    for (int r = 0; r < 8; ++r) {
      p0[r] = __expf(s0.f[r] - mnew);
      p1[r] = __expf(s1.f[r] - mnew);
      cs += p0[r] + p1[r];
    }
    cs += __shfl_xor(cs, 16, 32);
    lrun = lrun * alpha + cs;
    mrun = mnew;
    // --- rescale O (broadcast alpha per accumulator row) ---------------
    float ar[8];
#pragma unroll
    for (int r = 0; r < 8; ++r) ar[r] = __shfl(alpha, r + qsel, 32);
#pragma unroll
    for (int n = 0; n < 8; ++n)
#pragma unroll
      for (int r = 0; r < 8; ++r) O[n].f[r] *= ar[r];
    // --- pack P into A-fragment (layout matches C-fragment pair) -------
    Frag P;
#pragma unroll
    for (int v = 0; v < 4; ++v) {
      P.u[v]     = pack_bf2(p0[2 * v], p0[2 * v + 1]);
      P.u[4 + v] = pack_bf2(p1[2 * v], p1[2 * v + 1]);
    }
    // --- O += P * V (V^T rows are contiguous) --------------------------
#pragma unroll
    for (int n = 0; n < 8; ++n) {
      Frag vf; load_frag(vf, vp + (size_t)(n * 16 + r15) * S_ + kb + ko);
      O[n].v = WMMA_BF16(P.v, vf.v, O[n].v);
    }
  }

  // --- finalize: divide by l, store to [B,S,H*D] bf16 ------------------
  float lr[8];
#pragma unroll
  for (int r = 0; r < 8; ++r) lr[r] = 1.0f / __shfl(lrun, r + qsel, 32);
#pragma unroll
  for (int n = 0; n < 8; ++n) {
    const int dd = n * 16 + r15;
#pragma unroll
    for (int r = 0; r < 8; ++r) {
      const int qq = q0 + r + qsel;
      Ob[((size_t)(b * S_ + qq)) * E_ + h * D_ + dd] = f2bf(O[n].f[r] * lr[r]);
    }
  }
}

// ---------------------------------------------------------------------------
// Kernel 3: output projection, f32 result to d_out
// ---------------------------------------------------------------------------
__global__ __launch_bounds__(128)
void outproj_kernel(const uint16_t* __restrict__ Ain,  // [M_, E_] bf16
                    const uint16_t* __restrict__ W,    // [E_, E_] bf16
                    const float* __restrict__ bias,    // [E_]
                    float* __restrict__ out) {         // [M_, E_] f32
  const int lane = threadIdx.x & 31;
  const int wave = threadIdx.x >> 5;
  const int bm = blockIdx.x & 31;          // M_/128 = 32
  const int bn = blockIdx.x >> 5;          // E_/128 = 16
  const int mbase = bm * 128 + (wave & 1) * 64;
  const int nbase = bn * 128 + (wave >> 1) * 64;

  f32x8 acc[16];
  gemm64x64(Ain, E_, W, E_, E_, mbase, nbase, lane, acc);

  const int r15 = lane & 15;
  const int mo  = (lane < 16) ? 0 : 8;
#pragma unroll
  for (int j = 0; j < 4; ++j) {
    const int ncol = nbase + 16 * j + r15;
    const float bj = bias[ncol];
#pragma unroll
    for (int i = 0; i < 4; ++i) {
      FAcc fa; fa.v = acc[i * 4 + j];
#pragma unroll
      for (int r = 0; r < 8; ++r) {
        const int mm = mbase + 16 * i + r + mo;
        out[(size_t)mm * E_ + ncol] = fa.f[r] + bj;
      }
    }
  }
}

// ---------------------------------------------------------------------------
// Launch
// ---------------------------------------------------------------------------
extern "C" void kernel_launch(void* const* d_in, const int* in_sizes, int n_in,
                              void* d_out, int out_size, void* d_ws, size_t ws_size,
                              hipStream_t stream) {
  const float* x      = (const float*)d_in[0];
  // d_in[1] = attn_mask (causal, hardcoded)
  const float* w_qkv  = (const float*)d_in[2];
  const float* b_qkv  = (const float*)d_in[3];
  const float* w_out  = (const float*)d_in[4];
  const float* b_out  = (const float*)d_in[5];
  float* out          = (float*)d_out;

  char* ws = (char*)d_ws;
  size_t off = 0;
  auto wsalloc = [&](size_t bytes) -> void* {
    void* p = ws + off;
    off += (bytes + 255) & ~(size_t)255;
    return p;
  };
  uint16_t* xb    = (uint16_t*)wsalloc((size_t)M_ * E_ * 2);   // x bf16
  uint16_t* wqb   = (uint16_t*)wsalloc((size_t)N3_ * E_ * 2);  // w_qkv bf16
  uint16_t* wob   = (uint16_t*)wsalloc((size_t)E_ * E_ * 2);   // w_out bf16
  uint16_t* qbuf  = (uint16_t*)wsalloc((size_t)M_ * E_ * 2);   // Q [B,H,S,D]
  uint16_t* kbuf  = (uint16_t*)wsalloc((size_t)M_ * E_ * 2);   // K [B,H,S,D]
  uint16_t* vtbuf = (uint16_t*)wsalloc((size_t)M_ * E_ * 2);   // V^T [B,H,D,S]
  uint16_t* abuf  = (uint16_t*)wsalloc((size_t)M_ * E_ * 2);   // attn out [M,E]

  const int n_x  = M_ * E_;     // 8388608
  const int n_wq = N3_ * E_;    // 12582912
  const int n_wo = E_ * E_;     // 4194304
  cvt_f32_bf16<<<n_x  / 1024, 256, 0, stream>>>(x,     xb,  n_x);
  cvt_f32_bf16<<<n_wq / 1024, 256, 0, stream>>>(w_qkv, wqb, n_wq);
  cvt_f32_bf16<<<n_wo / 1024, 256, 0, stream>>>(w_out, wob, n_wo);

  qkv_gemm_kernel<<<(M_ / 128) * (N3_ / 128), 128, 0, stream>>>(
      xb, wqb, b_qkv, qbuf, kbuf, vtbuf);

  attn_kernel<<<B_ * H_ * (S_ / 128), 256, 0, stream>>>(qbuf, kbuf, vtbuf, abuf);

  outproj_kernel<<<(M_ / 128) * (E_ / 128), 128, 0, stream>>>(abuf, wob, b_out, out);
}
